// TT_Linear2_21182778704891
// MI455X (gfx1250) — matrix operand order repaired and checked by reading, not compiled, not run on hardware
//
#include <hip/hip_runtime.h>
#include <hip/hip_bf16.h>

// ---------------------------------------------------------------------------
// TT-linear on MI455X (gfx1250):
//   K0: convert X f32 -> bf16 (one pass, memory bound).
//   K1: build W^T (bf16, 4096x4096) from the 4 TT cores, all stages in LDS.
//   K2: out = X * W + bias; bf16 WMMA GEMM, f32 accumulate.
//       128x128x32 tiles, 8 wave32 waves (2x4), double-buffered LDS fed by
//       CDNA5 async DMA (global_load_async_to_lds_b128 / s_wait_asynccnt):
//       no register staging, no ds_store phase, minimal VGPR pressure.
// ---------------------------------------------------------------------------

typedef __attribute__((ext_vector_type(16))) __bf16 bf16x16;
typedef __attribute__((ext_vector_type(8)))  __bf16 bf16x8;
typedef __attribute__((ext_vector_type(8)))  float  floatx8;

#define BATCH    8192
#define IN_FEAT  4096
#define OUT_FEAT 4096

// float -> bf16 bits, round-to-nearest-even
__device__ __forceinline__ unsigned short f2bf(float f) {
  unsigned int u = __float_as_uint(f);
  u += 0x7FFFu + ((u >> 16) & 1u);
  return (unsigned short)(u >> 16);
}

// ---------------------------------------------------------------------------
// Kernel 0: X (f32) -> Xbf (bf16 bits), 8 elements per thread.
// ---------------------------------------------------------------------------
__global__ __launch_bounds__(256) void x_to_bf16(
    const float* __restrict__ X, unsigned short* __restrict__ Xbf)
{
  const size_t i = ((size_t)blockIdx.x * 256 + threadIdx.x) * 8;
  const float4 a = *(const float4*)(X + i);
  const float4 b = *(const float4*)(X + i + 4);
  uint4 o;
  o.x = (unsigned)f2bf(a.x) | ((unsigned)f2bf(a.y) << 16);
  o.y = (unsigned)f2bf(a.z) | ((unsigned)f2bf(a.w) << 16);
  o.z = (unsigned)f2bf(b.x) | ((unsigned)f2bf(b.y) << 16);
  o.w = (unsigned)f2bf(b.z) | ((unsigned)f2bf(b.w) << 16);
  *(uint4*)(Xbf + i) = o;
}

// ---------------------------------------------------------------------------
// Kernel 1: Wt[o][i] = sum_{r1,r2,r3} c0[i0,o0,r1] c1[r1,i1,o1,r2]
//                                     c2[r2,i2,o2,r3] c3[r3,i3,o3]
// One block per (i0,o0,i1,o1); T1/T2/core3 staged in LDS; bf16 output, transposed.
// ---------------------------------------------------------------------------
__global__ __launch_bounds__(256) void tt_build_wt(
    const float* __restrict__ c0, const float* __restrict__ c1,
    const float* __restrict__ c2, const float* __restrict__ c3,
    unsigned short* __restrict__ Wt)
{
  const int bid = blockIdx.x;
  const int o1 = bid & 7, i1 = (bid >> 3) & 7;
  const int o0 = (bid >> 6) & 7, i0 = (bid >> 9) & 7;
  const int tid = threadIdx.x;

  __shared__ float t1[32];
  __shared__ float t2[2048];
  __shared__ float c3s[2048];

  if (tid < 32) {
    const int r2 = tid;
    float s = 0.f;
#pragma unroll
    for (int r1 = 0; r1 < 32; ++r1)
      s += c0[(i0 * 8 + o0) * 32 + r1] * c1[((r1 * 8 + i1) * 8 + o1) * 32 + r2];
    t1[r2] = s;
  }
#pragma unroll
  for (int j = tid; j < 2048; j += 256) c3s[j] = c3[j];
  __syncthreads();

#pragma unroll
  for (int j = tid; j < 2048; j += 256) {
    const int r3 = j & 31, o2 = (j >> 5) & 7, i2 = j >> 8;
    float s = 0.f;
#pragma unroll
    for (int r2 = 0; r2 < 32; ++r2)
      s += t1[r2] * c2[((r2 * 8 + i2) * 8 + o2) * 32 + r3];
    t2[j] = s;
  }
  __syncthreads();

#pragma unroll
  for (int j = tid; j < 4096; j += 256) {
    const int o3 = j & 7, i3 = (j >> 3) & 7, o2 = (j >> 6) & 7, i2 = j >> 9;
    float s = 0.f;
#pragma unroll
    for (int r3 = 0; r3 < 32; ++r3)
      s += t2[(i2 * 8 + o2) * 32 + r3] * c3s[(r3 * 8 + i3) * 8 + o3];
    const int o = ((o0 * 8 + o1) * 8 + o2) * 8 + o3;
    const int i = ((i0 * 8 + i1) * 8 + i2) * 8 + i3;
    Wt[(size_t)o * IN_FEAT + i] = f2bf(s);
  }
}

// ---------------------------------------------------------------------------
// Kernel 2: out[m][n] = sum_k Xbf[m][k] * Wt[n][k] + bias[n]
// ---------------------------------------------------------------------------
#define BM 128
#define BN 128
#define BK 32
#define LDT 40                       // padded LDS row stride (bf16 elems), 80 B
#define TILE_BYTES (BM * LDT * 2)    // 10240 B per buffer

__global__ __launch_bounds__(256) void tt_gemm_bias(
    const unsigned short* __restrict__ Xbf, const unsigned short* __restrict__ Wt,
    const float* __restrict__ bias, float* __restrict__ out)
{
  __shared__ __align__(16) unsigned short Ash[2][BM][LDT];
  __shared__ __align__(16) unsigned short Bsh[2][BN][LDT];

  const int tid   = threadIdx.x;
  const int lane  = tid & 31;
  const int wave  = tid >> 5;       // 0..7 (wave32)
  const int wm    = wave & 1;       // 2 waves along M
  const int wn    = wave >> 1;      // 4 waves along N
  const int l15   = lane & 15;
  const int mhalf = lane >> 4;
  const int m0 = blockIdx.y * BM;
  const int n0 = blockIdx.x * BN;

  // Tile feed: 128 rows x 32 bf16 = 512 x 16B chunks per matrix; 2 per thread.
  const int row = tid >> 2;         // q adds 64
  const int seg = tid & 3;          // 16B chunk within row

  // Per-thread global bases (advance by kt*BK elements at issue time).
  const unsigned short* aG[2];
  const unsigned short* bG[2];
  unsigned aL[2], bL[2];            // LDS byte addresses (buffer 0)
#pragma unroll
  for (int q = 0; q < 2; ++q) {
    aG[q] = Xbf + (size_t)(m0 + row + q * 64) * IN_FEAT + seg * 8;
    bG[q] = Wt  + (size_t)(n0 + row + q * 64) * IN_FEAT + seg * 8;
    aL[q] = (unsigned)(unsigned long long)&Ash[0][row + q * 64][seg * 8];
    bL[q] = (unsigned)(unsigned long long)&Bsh[0][row + q * 64][seg * 8];
  }

  // Async DMA one K-tile into LDS buffer `buf` (4 in-order async ops/thread).
  auto issueTile = [&](int kt, int buf) {
    const size_t ko = (size_t)kt * BK;
    const unsigned bofs = buf ? TILE_BYTES : 0u;
#pragma unroll
    for (int q = 0; q < 2; ++q) {
      const unsigned short* ga = aG[q] + ko;
      const unsigned short* gb = bG[q] + ko;
      unsigned la = aL[q] + bofs;
      unsigned lb = bL[q] + bofs;
      asm volatile("global_load_async_to_lds_b128 %0, %1, off"
                   :: "v"(la), "v"(ga) : "memory");
      asm volatile("global_load_async_to_lds_b128 %0, %1, off"
                   :: "v"(lb), "v"(gb) : "memory");
    }
  };

  floatx8 acc[4][2];
#pragma unroll
  for (int a = 0; a < 4; ++a)
#pragma unroll
    for (int b = 0; b < 2; ++b) acc[a][b] = (floatx8)0.f;

  const int KT = IN_FEAT / BK;      // 128
  issueTile(0, 0);

  for (int kt = 0; kt < KT; ++kt) {
    const int buf = kt & 1;
    if (kt + 1 < KT) {
      issueTile(kt + 1, buf ^ 1);
      asm volatile("s_wait_asynccnt 4" ::: "memory");   // tile kt complete
    } else {
      asm volatile("s_wait_asynccnt 0" ::: "memory");
    }
    __syncthreads();

    // A fragment (16x32 bf16): lane<16 -> K {0-7,16-23}; lane>=16 -> {8-15,24-31}
    bf16x16 afr[4];
#pragma unroll
    for (int a = 0; a < 4; ++a) {
      const unsigned short* rp = &Ash[buf][wm * 64 + a * 16 + l15][0];
      bf16x8 lo = *(const bf16x8*)(rp + mhalf * 8);
      bf16x8 hi = *(const bf16x8*)(rp + mhalf * 8 + 16);
      bf16x16 v;
#pragma unroll
      for (int e = 0; e < 8; ++e) { v[e] = lo[e]; v[e + 8] = hi[e]; }
      afr[a] = v;
    }
    // B fragment (32x16 bf16): lane<16 -> K {0-7,16-23}; lane>=16 -> {8-15,24-31}
    bf16x16 bfr[2];
#pragma unroll
    for (int b = 0; b < 2; ++b) {
      const unsigned short* rp = &Bsh[buf][wn * 32 + b * 16 + l15][mhalf * 8];
      bf16x8 lo = *(const bf16x8*)(rp);
      bf16x8 hi = *(const bf16x8*)(rp + 16);
      bf16x16 v;
#pragma unroll
      for (int e = 0; e < 8; ++e) { v[e] = lo[e]; v[e + 8] = hi[e]; }
      bfr[b] = v;
    }

#pragma unroll
    for (int a = 0; a < 4; ++a)
#pragma unroll
      for (int b = 0; b < 2; ++b)
        acc[a][b] = __builtin_amdgcn_wmma_f32_16x16x32_bf16(
            false, afr[a], false, bfr[b], (short)0, acc[a][b], false, false);

    __syncthreads();
  }

  // C/D layout: VGPR r, lanes 0-15 -> (M=r, N=lane); lanes 16-31 -> (M=r+8).
#pragma unroll
  for (int b = 0; b < 2; ++b) {
    const int col = n0 + wn * 32 + b * 16 + l15;
    const float bv = bias[col];
#pragma unroll
    for (int a = 0; a < 4; ++a) {
      const int rbase = m0 + wm * 64 + a * 16 + mhalf * 8;
#pragma unroll
      for (int r = 0; r < 8; ++r)
        out[(size_t)(rbase + r) * OUT_FEAT + col] = acc[a][b][r] + bv;
    }
  }
}

// ---------------------------------------------------------------------------
extern "C" void kernel_launch(void* const* d_in, const int* in_sizes, int n_in,
                              void* d_out, int out_size, void* d_ws, size_t ws_size,
                              hipStream_t stream) {
  const float* x    = (const float*)d_in[0];
  const float* c0   = (const float*)d_in[1];
  const float* c1   = (const float*)d_in[2];
  const float* c2   = (const float*)d_in[3];
  const float* c3   = (const float*)d_in[4];
  const float* bias = (const float*)d_in[5];
  float* out = (float*)d_out;

  unsigned short* Wt  = (unsigned short*)d_ws;                       // 32 MB
  unsigned short* Xbf = (unsigned short*)((char*)d_ws + (size_t)OUT_FEAT * IN_FEAT * 2); // 64 MB

  x_to_bf16<<<(BATCH * IN_FEAT) / (256 * 8), 256, 0, stream>>>(x, Xbf);
  tt_build_wt<<<4096, 256, 0, stream>>>(c0, c1, c2, c3, Wt);

  dim3 grid(OUT_FEAT / BN, BATCH / BM);  // (32, 64)
  tt_gemm_bias<<<grid, 256, 0, stream>>>(Xbf, Wt, bias, out);
}
